// CrossAttention_12403865551530
// MI455X (gfx1250) — matrix operand order repaired and checked by reading, not compiled, not run on hardware
//
#include <hip/hip_runtime.h>
#include <math.h>

typedef __attribute__((ext_vector_type(16))) __bf16          v16bf;
typedef __attribute__((ext_vector_type(8)))  float           v8f;
typedef __attribute__((ext_vector_type(8)))  unsigned short  ushort8;

union Frag {
    unsigned short u[16];
    ushort8        h[2];
    v16bf          v;
};

__device__ __forceinline__ unsigned short f2bf(float f) {
    unsigned int u = __float_as_uint(f);
    unsigned int r = u + 0x7FFFu + ((u >> 16) & 1u);   // round-to-nearest-even
    return (unsigned short)(r >> 16);
}

__device__ __forceinline__ v8f wmma_bf16(const Frag& a, const Frag& b, v8f c) {
    return __builtin_amdgcn_wmma_f32_16x16x32_bf16(
        false, a.v, false, b.v, (short)0, c, false, false);
}

// ---------------------------------------------------------------- converts
__global__ __launch_bounds__(256) void cvt_bf16_kernel(
    const float* __restrict__ in, unsigned short* __restrict__ out, int n)
{
    int i = blockIdx.x * 256 + threadIdx.x;
    if (i < n) out[i] = f2bf(in[i]);
}

__global__ __launch_bounds__(256) void cvt_wT_kernel(
    const float* __restrict__ Wq, const float* __restrict__ Wk,
    const float* __restrict__ Wv,
    unsigned short* __restrict__ qT, unsigned short* __restrict__ kT,
    unsigned short* __restrict__ vT, int D)
{
    int i = blockIdx.x * 256 + threadIdx.x;
    if (i < D * D) {
        int n = i / D, k = i % D;                    // WT[n][k] = W[k][n]
        size_t src = (size_t)k * D + n;
        qT[i] = f2bf(Wq[src]);
        kT[i] = f2bf(Wk[src]);
        vT[i] = f2bf(Wv[src]);
    }
}

// ---------------------------------------------------------------- QKV GEMM
// out[m,n] = sum_k A[m,k]*W[k,n] + bias[n];  A bf16 [M,D], WT bf16 [n-major].
// block = 256 thr (8 waves) covers 32 rows x 256 cols; wave tile 16x64.
// Software-pipelined: fragments for K-step k+32 are in flight during the
// WMMAs of step k, so waits are partial instead of full loadcnt drains.
__global__ __launch_bounds__(256) void qkv_gemm_kernel(
    const unsigned short* __restrict__ A,
    const unsigned short* __restrict__ WT,
    const float* __restrict__ bias,
    unsigned short* __restrict__ out,
    int M, int D, int S, int transpose_out)
{
    const int lane = threadIdx.x & 31;
    const int wave = threadIdx.x >> 5;
    const int hl   = lane >> 4;
    const int ln16 = lane & 15;
    const int m0   = blockIdx.x * 32 + (wave >> 2) * 16;
    const int n0   = blockIdx.y * 256 + (wave & 3) * 64;

    v8f acc[4] = {};
    const unsigned short* arow = A + (size_t)(m0 + ln16) * D;
    const unsigned short* brow[4];
#pragma unroll
    for (int t = 0; t < 4; ++t)
        brow[t] = WT + (size_t)(n0 + t * 16 + ln16) * D;

    auto loadA = [&](Frag& f, int k0) {
        f.h[0] = *(const ushort8*)(arow + k0 + hl * 8);
        f.h[1] = *(const ushort8*)(arow + k0 + 16 + hl * 8);
    };
    auto loadB = [&](Frag* f, int k0) {
#pragma unroll
        for (int t = 0; t < 4; ++t) {
            f[t].h[0] = *(const ushort8*)(brow[t] + k0 + hl * 8);
            f[t].h[1] = *(const ushort8*)(brow[t] + k0 + 16 + hl * 8);
        }
    };

    Frag a0, a1, b0[4], b1[4];
    loadA(a0, 0);
    loadB(b0, 0);
    for (int k0 = 0; k0 < D; k0 += 64) {            // D % 64 == 0
        __builtin_prefetch(arow + k0 + 128, 0, 3);
        loadA(a1, k0 + 32);
        loadB(b1, k0 + 32);
#pragma unroll
        for (int t = 0; t < 4; ++t) acc[t] = wmma_bf16(a0, b0[t], acc[t]);
        if (k0 + 64 < D) { loadA(a0, k0 + 64); loadB(b0, k0 + 64); }
#pragma unroll
        for (int t = 0; t < 4; ++t) acc[t] = wmma_bf16(a1, b1[t], acc[t]);
    }

#pragma unroll
    for (int t = 0; t < 4; ++t) {
        int n = n0 + t * 16 + ln16;
        float bs = bias[n];
#pragma unroll
        for (int r = 0; r < 8; ++r) {
            int m = m0 + r + hl * 8;                 // C layout: vgpr r -> M=r / r+8
            unsigned short bv = f2bf(acc[t][r] + bs);
            if (transpose_out) {                      // vT[b][n][s]
                int bb = m / S, s = m - bb * S;
                out[((size_t)bb * D + n) * S + s] = bv;
            } else {
                out[(size_t)m * D + n] = bv;
            }
        }
    }
}

// -------------------------------------------- fused scores + segment softmax
// One block per (b, 16-query tile). 8 waves each compute a 16x256 score slice
// (16 accumulators, pipelined B loads) into LDS; then 256 threads
// (16 rows x 16 patches) do the segment softmax and write bf16 probabilities.
#define SROW 2049   // padded LDS row stride (floats)

__global__ __launch_bounds__(256) void scores_softmax_kernel(
    const unsigned short* __restrict__ qb,
    const unsigned short* __restrict__ kb,
    const long long* __restrict__ pidx,
    unsigned short* __restrict__ Pmat,
    int S, int D, int Pp, float scale)
{
    extern __shared__ float sc[];                  // [16][SROW]
    const int qt   = blockIdx.x % (S / 16);
    const int b    = blockIdx.x / (S / 16);
    const int q0   = qt * 16;
    const int lane = threadIdx.x & 31;
    const int wave = threadIdx.x >> 5;
    const int hl   = lane >> 4;
    const int ln16 = lane & 15;
    const int kw0  = wave * 256;                   // this wave's key range

    v8f acc[16] = {};
    const unsigned short* qrow  = qb + ((size_t)b * S + q0 + ln16) * D;
    const unsigned short* kbase = kb + ((size_t)b * S + kw0 + ln16) * D;

    auto loadA = [&](Frag& f, int d0) {
        f.h[0] = *(const ushort8*)(qrow + d0 + hl * 8);
        f.h[1] = *(const ushort8*)(qrow + d0 + 16 + hl * 8);
    };
    // one K=32 slice of the score tile, with ping-pong B fragments
    auto substep = [&](const Frag& a, int d0) {
        const unsigned short* kp = kbase + d0 + hl * 8;
        Frag p0, p1;
        p0.h[0] = *(const ushort8*)kp;
        p0.h[1] = *(const ushort8*)(kp + 16);
#pragma unroll
        for (int kt = 0; kt < 16; kt += 2) {
            const unsigned short* kp1 = kp + (size_t)(kt + 1) * 16 * D;
            p1.h[0] = *(const ushort8*)kp1;
            p1.h[1] = *(const ushort8*)(kp1 + 16);
            acc[kt] = wmma_bf16(a, p0, acc[kt]);
            if (kt + 2 < 16) {
                const unsigned short* kp2 = kp + (size_t)(kt + 2) * 16 * D;
                p0.h[0] = *(const ushort8*)kp2;
                p0.h[1] = *(const ushort8*)(kp2 + 16);
            }
            acc[kt + 1] = wmma_bf16(a, p1, acc[kt + 1]);
        }
    };

    Frag a0, a1;
    loadA(a0, 0);
    for (int d0 = 0; d0 < D; d0 += 64) {
        __builtin_prefetch(qrow + d0 + 128, 0, 3);
        loadA(a1, d0 + 32);
        substep(a0, d0);
        if (d0 + 64 < D) loadA(a0, d0 + 64);
        substep(a1, d0 + 32);
    }

#pragma unroll
    for (int kt = 0; kt < 16; ++kt)
#pragma unroll
        for (int r = 0; r < 8; ++r)
            sc[(size_t)(r + hl * 8) * SROW + kw0 + kt * 16 + ln16] =
                acc[kt][r] * scale;

    __syncthreads();

    // segment softmax: thread -> (row r, patch p); exp stored back into LDS
    const int r = threadIdx.x & 15;
    const int p = threadIdx.x >> 4;
    int ks = (int)pidx[b * Pp + p];
    int ke = (p == Pp - 1) ? S : (int)pidx[b * Pp + p + 1];

    float* srow = sc + (size_t)r * SROW;
    float mx = -3.0e38f;
    for (int k = ks; k < ke; ++k) mx = fmaxf(mx, srow[k]);
    float sum = 0.f;
    for (int k = ks; k < ke; ++k) {
        float e = __expf(srow[k] - mx);
        srow[k] = e;
        sum += e;
    }
    float inv = 1.0f / sum;

    unsigned short* prow = Pmat + ((size_t)b * S + q0 + r) * S;
    for (int k = ks; k < ke; ++k)
        prow[k] = f2bf(srow[k] * inv);
}

// ---------------------------------------------------------------- P @ V
// out[b,p,q,d] = sum_{k in [ks,ke)} Pmat[b,q,k] * v[b,k,d]; vT is [B,D,S].
// K loop is aligned down to a 32 boundary so interior tiles use b128 loads;
// only the first/last tiles take the masked path (ks..ke are block-uniform,
// so the branch is scalar). Leading elements k<ks MUST be masked: they hold
// the previous segment's probabilities.
__global__ __launch_bounds__(256) void pv_gemm_kernel(
    const unsigned short* __restrict__ Pmat,
    const unsigned short* __restrict__ vT,
    const long long* __restrict__ pidx,
    float* __restrict__ out,
    int S, int D, int Pp)
{
    const int b  = blockIdx.x / Pp;
    const int p  = blockIdx.x % Pp;
    const int ks = (int)pidx[b * Pp + p];
    const int ke = (p == Pp - 1) ? S : (int)pidx[b * Pp + p + 1];

    const int lane = threadIdx.x & 31;
    const int wave = threadIdx.x >> 5;
    const int hl   = lane >> 4;
    const int ln16 = lane & 15;
    const int q0   = blockIdx.y * 32 + (wave >> 2) * 16;
    const int d0   = blockIdx.z * 256 + (wave & 3) * 64;

    v8f acc[4] = {};
    const unsigned short* prow = Pmat + ((size_t)b * S + q0 + ln16) * S;
    const unsigned short* vbase[4];
#pragma unroll
    for (int t = 0; t < 4; ++t)
        vbase[t] = vT + ((size_t)b * D + d0 + t * 16 + ln16) * S;

    const int k0a = ks & ~31;                       // 32-aligned loop base
    for (int kb0 = k0a; kb0 < ke; kb0 += 32) {
        const bool full = (kb0 >= ks) && (kb0 + 32 <= ke);   // uniform branch
        __builtin_prefetch(prow + kb0 + 32, 0, 3);
        Frag a;
        if (full) {
            a.h[0] = *(const ushort8*)(prow + kb0 + hl * 8);
            a.h[1] = *(const ushort8*)(prow + kb0 + 16 + hl * 8);
        } else {
#pragma unroll
            for (int j = 0; j < 8; ++j) {
                int k1 = kb0 + hl * 8 + j;
                a.u[j] = (k1 >= ks && k1 < ke) ? prow[k1] : (unsigned short)0;
                int k2 = kb0 + 16 + hl * 8 + j;
                a.u[8 + j] = (k2 >= ks && k2 < ke) ? prow[k2] : (unsigned short)0;
            }
        }
#pragma unroll
        for (int t = 0; t < 4; ++t) {
            Frag bfr;                                // B[kk, n] = v[kb0+kk, d]
            if (full) {
                bfr.h[0] = *(const ushort8*)(vbase[t] + kb0 + hl * 8);
                bfr.h[1] = *(const ushort8*)(vbase[t] + kb0 + 16 + hl * 8);
            } else {
#pragma unroll
                for (int j = 0; j < 16; ++j) {
                    int k = kb0 + hl * 8 + (j & 7) + ((j >> 3) << 4);
                    bfr.u[j] = (k >= ks && k < ke) ? vbase[t][k]
                                                   : (unsigned short)0;
                }
            }
            acc[t] = wmma_bf16(a, bfr, acc[t]);
        }
    }

#pragma unroll
    for (int t = 0; t < 4; ++t)
#pragma unroll
        for (int r = 0; r < 8; ++r) {
            int q = q0 + r + hl * 8;
            int d = d0 + t * 16 + ln16;
            out[(((size_t)b * Pp + p) * S + q) * D + d] = acc[t][r];
        }
}

// ---------------------------------------------------------------- launch
extern "C" void kernel_launch(void* const* d_in, const int* in_sizes, int n_in,
                              void* d_out, int out_size, void* d_ws, size_t ws_size,
                              hipStream_t stream)
{
    (void)in_sizes; (void)n_in; (void)out_size; (void)ws_size;
    const int B = 2, S = 2048, D = 768, P = 16;
    const float scale = 1.0f / sqrtf((float)D);

    const float*     x    = (const float*)d_in[0];
    const long long* pidx = (const long long*)d_in[1];   // int64 per reference
    const float*     Wq   = (const float*)d_in[2];
    const float*     bq   = (const float*)d_in[3];
    const float*     Wk   = (const float*)d_in[4];
    const float*     bk   = (const float*)d_in[5];
    const float*     Wv   = (const float*)d_in[6];
    const float*     bv   = (const float*)d_in[7];
    float*           out  = (float*)d_out;

    char* w = (char*)d_ws;
    auto au = [](size_t v) { return (v + 255) & ~(size_t)255; };
    size_t off = 0;
    unsigned short* xb  = (unsigned short*)(w + off); off += au((size_t)B * S * D * 2);
    unsigned short* wqT = (unsigned short*)(w + off); off += au((size_t)D * D * 2);
    unsigned short* wkT = (unsigned short*)(w + off); off += au((size_t)D * D * 2);
    unsigned short* wvT = (unsigned short*)(w + off); off += au((size_t)D * D * 2);
    unsigned short* qb  = (unsigned short*)(w + off); off += au((size_t)B * S * D * 2);
    unsigned short* kb  = (unsigned short*)(w + off); off += au((size_t)B * S * D * 2);
    unsigned short* vT  = (unsigned short*)(w + off); off += au((size_t)B * S * D * 2);
    unsigned short* Pm  = (unsigned short*)(w + off); off += au((size_t)B * S * S * 2);

    // 1) bf16 conversions
    cvt_bf16_kernel<<<(B * S * D + 255) / 256, 256, 0, stream>>>(x, xb, B * S * D);
    cvt_wT_kernel<<<(D * D + 255) / 256, 256, 0, stream>>>(Wq, Wk, Wv, wqT, wkT, wvT, D);

    // 2) projections (M = B*S rows); v stored transposed [B,D,S]
    dim3 gq((B * S) / 32, D / 256);
    qkv_gemm_kernel<<<gq, 256, 0, stream>>>(xb, wqT, bq, qb, B * S, D, S, 0);
    qkv_gemm_kernel<<<gq, 256, 0, stream>>>(xb, wkT, bk, kb, B * S, D, S, 0);
    qkv_gemm_kernel<<<gq, 256, 0, stream>>>(xb, wvT, bv, vT, B * S, D, S, 1);

    // 3) fused scores + per-segment softmax -> bf16 probabilities
    size_t lds_bytes = (size_t)16 * SROW * sizeof(float);   // 128 KB + pad
    scores_softmax_kernel<<<B * (S / 16), 256, lds_bytes, stream>>>(
        qb, kb, pidx, Pm, S, D, P, scale);

    // 4) P @ V -> fp32 output [B,P,S,D]
    dim3 gpv(B * P, S / 32, D / 256);
    pv_gemm_kernel<<<gpv, 256, 0, stream>>>(Pm, vT, pidx, out, S, D, P);
}
